// LSTM_3100966387953
// MI455X (gfx1250) — compile-verified, matrix operand order checked by reading, not run on hardware
//
#include <hip/hip_runtime.h>
#include <hip/hip_bf16.h>

// ---------------------------------------------------------------------------
// LSTM  (T=512, B=64, H=E=1024)  for MI455X / gfx1250  (wave32, bf16 WMMA)
//
// Phase 1: all input projections as one big bf16 WMMA GEMM (parallel over T).
// Phase 2: ONE persistent kernel runs the 512-step recurrence. Producer/
//          consumer flag sync (group-local, 32 flags) instead of a grid-wide
//          barrier; h tile staged into LDS via global_load_async_to_lds_b128
//          (ASYNCcnt); weight fragments software-pipelined from L2.
// ---------------------------------------------------------------------------

#define T_STEPS 512
#define BATCH   64
#define HDIM    1024
#define EDIM    1024
#define GDIM    4096   /* 4*HDIM */
#define RBLK    128    /* persistent blocks: 4 m-tiles x 32 n-groups */

typedef __attribute__((ext_vector_type(8)))  __bf16 v8bf;
typedef __attribute__((ext_vector_type(16))) __bf16 v16bf;
typedef __attribute__((ext_vector_type(8)))  float  v8f;

// float -> bf16 (round-to-nearest-even) as raw bits
__device__ __forceinline__ unsigned short f2bf(float x) {
    unsigned int u = __float_as_uint(x);
    u = (u + 0x7FFFu + ((u >> 16) & 1u)) >> 16;
    return (unsigned short)u;
}

// One lane's slice of a 16x32 / 32x16 bf16 WMMA fragment (ISA 7.12.2):
// two 16B chunks at K..K+7 and K+16..K+23. Works for global and LDS
// pointers (addrspace inferred -> global_load_b128 / ds_load_b128).
__device__ __forceinline__ v16bf ld_frag(const unsigned short* p) {
    v8bf lo = *(const v8bf*)(p);
    v8bf hi = *(const v8bf*)(p + 16);
    return __builtin_shufflevector(lo, hi, 0,1,2,3,4,5,6,7,8,9,10,11,12,13,14,15);
}

__device__ __forceinline__ float sig_fast(float x) {
    return __builtin_amdgcn_rcpf(1.0f + __expf(-x));
}
__device__ __forceinline__ float tanh_fast(float x) {
    return 2.0f * sig_fast(2.0f * x) - 1.0f;
}

#define WMMA_BF16(a, b, c) \
    __builtin_amdgcn_wmma_f32_16x16x32_bf16(false, (a), false, (b), (short)0, (c), false, false)

// ---------------------------------------------------------------------------
// Pack weights (fp32 -> bf16, gate-major [4H, K] row-major) + pack biases.
// ---------------------------------------------------------------------------
__global__ void pack_params(const float* __restrict__ Wgx, const float* __restrict__ Wgh,
                            const float* __restrict__ Wix, const float* __restrict__ Wih,
                            const float* __restrict__ Wfx, const float* __restrict__ Wfh,
                            const float* __restrict__ Wox, const float* __restrict__ Woh,
                            const float* __restrict__ bg,  const float* __restrict__ bi,
                            const float* __restrict__ bf_, const float* __restrict__ bo,
                            unsigned short* __restrict__ Wx_p,
                            unsigned short* __restrict__ Wh_p,
                            float* __restrict__ bias_p) {
    const size_t NWX = (size_t)GDIM * EDIM;
    const size_t NWH = (size_t)GDIM * HDIM;
    size_t i0     = (size_t)blockIdx.x * blockDim.x + threadIdx.x;
    size_t stride = (size_t)gridDim.x * blockDim.x;
    for (size_t i = i0; i < NWX; i += stride) {
        int g = (int)(i / ((size_t)HDIM * EDIM));
        size_t r = i % ((size_t)HDIM * EDIM);
        float v = (g == 0) ? Wgx[r] : (g == 1) ? Wix[r] : (g == 2) ? Wfx[r] : Wox[r];
        Wx_p[i] = f2bf(v);
    }
    for (size_t i = i0; i < NWH; i += stride) {
        int g = (int)(i / ((size_t)HDIM * HDIM));
        size_t r = i % ((size_t)HDIM * HDIM);
        float v = (g == 0) ? Wgh[r] : (g == 1) ? Wih[r] : (g == 2) ? Wfh[r] : Woh[r];
        Wh_p[i] = f2bf(v);
    }
    for (size_t i = i0; i < (size_t)GDIM; i += stride) {
        int g = (int)(i / HDIM);
        int r = (int)(i % HDIM);
        bias_p[i] = (g == 0) ? bg[r] : (g == 1) ? bi[r] : (g == 2) ? bf_[r] : bo[r];
    }
}

__global__ void convert_embeds(const float* __restrict__ x, unsigned short* __restrict__ xb) {
    const size_t N = (size_t)T_STEPS * BATCH * EDIM;
    size_t i0     = (size_t)blockIdx.x * blockDim.x + threadIdx.x;
    size_t stride = (size_t)gridDim.x * blockDim.x;
    for (size_t i = i0; i < N; i += stride) xb[i] = f2bf(x[i]);
}

// Zero h0 (bf16), c0 (fp32) and the producer flags (every launch).
__global__ void init_state(unsigned short* __restrict__ h0, float* __restrict__ c0,
                           unsigned int* __restrict__ flags) {
    size_t i = (size_t)blockIdx.x * blockDim.x + threadIdx.x;
    if (i < (size_t)BATCH * HDIM) { h0[i] = 0; c0[i] = 0.0f; }
    if (i < (size_t)RBLK) flags[i] = 0u;
}

// ---------------------------------------------------------------------------
// Input-projection GEMM: Xpre[m,n] = sum_k Xbf[m,k]*Wx_p[n,k] + bias[n]
// M=32768, N=4096, K=1024. One wave: 16x64 strip, depth-1 register pipeline.
// grid = (64, 256), block = 256.
// ---------------------------------------------------------------------------
__global__ void __launch_bounds__(256)
xproj_gemm(const unsigned short* __restrict__ Xbf,
           const unsigned short* __restrict__ Wx_p,
           const float* __restrict__ bias_p,
           float* __restrict__ Xpre) {
    const int lane   = threadIdx.x & 31;
    const int wave   = threadIdx.x >> 5;
    const int m_tile = blockIdx.y * 8 + wave;
    const int n0     = blockIdx.x * 64;
    const int l15    = lane & 15;
    const int kb     = (lane >> 4) << 3;

    v8f acc0 = {}, acc1 = {}, acc2 = {}, acc3 = {};

    const unsigned short* a_row  = Xbf  + (size_t)(m_tile * 16 + l15) * EDIM + kb;
    const unsigned short* b_row0 = Wx_p + (size_t)(n0 +  0 + l15) * EDIM + kb;
    const unsigned short* b_row1 = Wx_p + (size_t)(n0 + 16 + l15) * EDIM + kb;
    const unsigned short* b_row2 = Wx_p + (size_t)(n0 + 32 + l15) * EDIM + kb;
    const unsigned short* b_row3 = Wx_p + (size_t)(n0 + 48 + l15) * EDIM + kb;

    v16bf a_c  = ld_frag(a_row);
    v16bf b0_c = ld_frag(b_row0), b1_c = ld_frag(b_row1);
    v16bf b2_c = ld_frag(b_row2), b3_c = ld_frag(b_row3);

#pragma unroll 4
    for (int k0 = 0; k0 < EDIM; k0 += 32) {
        int k1 = (k0 + 32 < EDIM) ? (k0 + 32) : 0;   // wrap: harmless extra load
        v16bf a_n  = ld_frag(a_row  + k1);
        v16bf b0_n = ld_frag(b_row0 + k1);
        v16bf b1_n = ld_frag(b_row1 + k1);
        v16bf b2_n = ld_frag(b_row2 + k1);
        v16bf b3_n = ld_frag(b_row3 + k1);
        acc0 = WMMA_BF16(a_c, b0_c, acc0);
        acc1 = WMMA_BF16(a_c, b1_c, acc1);
        acc2 = WMMA_BF16(a_c, b2_c, acc2);
        acc3 = WMMA_BF16(a_c, b3_c, acc3);
        a_c = a_n; b0_c = b0_n; b1_c = b1_n; b2_c = b2_n; b3_c = b3_n;
    }

    const int m_add = (lane >> 4) << 3;
#pragma unroll
    for (int v = 0; v < 8; ++v) {
        size_t m = (size_t)(m_tile * 16 + v + m_add);
        float* orow = Xpre + m * GDIM;
        int n = n0 + l15;
        orow[n +  0] = acc0[v] + bias_p[n +  0];
        orow[n + 16] = acc1[v] + bias_p[n + 16];
        orow[n + 32] = acc2[v] + bias_p[n + 32];
        orow[n + 48] = acc3[v] + bias_p[n + 48];
    }
}

// ---------------------------------------------------------------------------
// Persistent recurrence kernel: 128 blocks x 128 threads (4 waves).
//   block -> tile_m = blockIdx&3 (16 batch rows), n_grp = blockIdx>>2
//            (32 hidden cols). wave = gate; each wave computes two 16x16
//            gate-tiles (n_sub 0/1) sharing the A fragment (2 accumulators).
//   Sync: producer flags, group-local (only the 32 blocks sharing tile_m
//   produce the h rows this block consumes). No grid-wide barrier.
//   h tile staged into LDS with async global->LDS copies (ASYNCcnt).
// ---------------------------------------------------------------------------
__global__ void __launch_bounds__(128)
lstm_recurrence(unsigned short* __restrict__ hb0,
                unsigned short* __restrict__ hb1,
                float* __restrict__ c_st,
                const unsigned short* __restrict__ Wh_p,
                const float* __restrict__ Xpre,
                float* __restrict__ out,
                unsigned int* __restrict__ flags) {
    __shared__ unsigned short sA[16 * HDIM];     // 32 KB: h tile (16 rows x 1024)
    __shared__ float sPre[4][2][256];            //  8 KB: gate x n_sub preacts

    const int tid    = threadIdx.x;
    const int lane   = tid & 31;
    const int gate   = tid >> 5;                 // wave 0..3 = gate
    const int tile_m = blockIdx.x & 3;           // 0..3  (batch rows /16)
    const int n_grp  = blockIdx.x >> 2;          // 0..31 (hidden cols /32)
    const int l15    = lane & 15;
    const int kb     = (lane >> 4) << 3;
    const int m_add  = (lane >> 4) << 3;

    // weight rows for this wave's two gate-tiles (constant over t; L2-hot)
    const unsigned short* b0_row =
        Wh_p + (size_t)(gate * HDIM + n_grp * 32 +  0 + l15) * HDIM + kb;
    const unsigned short* b1_row =
        Wh_p + (size_t)(gate * HDIM + n_grp * 32 + 16 + l15) * HDIM + kb;
    const unsigned short* sa_row = sA + l15 * HDIM + kb;

    // base addresses for the async A-tile copy (this thread's 16B lanes)
    const unsigned int lds_base =
        (unsigned int)(unsigned long long)(uintptr_t)sA + (unsigned int)tid * 16u;
    unsigned int* const my_flag = &flags[blockIdx.x];
    unsigned int* const grp_flags = &flags[tile_m * 32];

    for (int t = 0; t < T_STEPS; ++t) {
        const unsigned short* h_in  = (t & 1) ? hb1 : hb0;
        unsigned short*       h_out = (t & 1) ? hb0 : hb1;

        // ---- wait for the 32 blocks 32*tile_m..32*tile_m+31 (step t-1) ----
        if (t > 0) {
            if (tid < 32) {
                while (__hip_atomic_load(&grp_flags[tid], __ATOMIC_ACQUIRE,
                                         __HIP_MEMORY_SCOPE_AGENT) < (unsigned)t) {
                    __builtin_amdgcn_s_sleep(1);
                }
            }
            __syncthreads();
        }

        // ---- stage h tile (16 x 1024 bf16 = 32 KB) into LDS, async ----
        {
            unsigned long long ga = (unsigned long long)(uintptr_t)
                (h_in + (size_t)tile_m * 16 * HDIM) + (unsigned long long)tid * 16ull;
#pragma unroll
            for (int i = 0; i < 16; ++i) {
                unsigned int la = lds_base + (unsigned int)(i * 2048);
                unsigned long long g = ga + (unsigned long long)(i * 2048);
                asm volatile("global_load_async_to_lds_b128 %0, %1, off"
                             :: "v"(la), "v"(g) : "memory");
            }
            asm volatile("s_wait_asynccnt 0x0" ::: "memory");
        }
        __syncthreads();

        // ---- GEMM: two 16x16 gate-tiles, A from LDS, B pipelined 2-deep ----
        v8f acc0 = {}, acc1 = {};
        {
            v16bf p00 = ld_frag(b0_row + 0),  p01 = ld_frag(b0_row + 32);
            v16bf p10 = ld_frag(b1_row + 0),  p11 = ld_frag(b1_row + 32);
#pragma unroll
            for (int kc = 0; kc < 32; ++kc) {
                int kn = (kc + 2 < 32) ? (kc + 2) * 32 : 0;  // wrap: harmless
                v16bf n0f = ld_frag(b0_row + kn);
                v16bf n1f = ld_frag(b1_row + kn);
                v16bf a   = ld_frag(sa_row + kc * 32);
                acc0 = WMMA_BF16(a, p00, acc0);
                acc1 = WMMA_BF16(a, p10, acc1);
                p00 = p01; p01 = n0f;
                p10 = p11; p11 = n1f;
            }
        }

        // ---- publish gate preacts to LDS ----
#pragma unroll
        for (int v = 0; v < 8; ++v) {
            sPre[gate][0][(v + m_add) * 16 + l15] = acc0[v];
            sPre[gate][1][(v + m_add) * 16 + l15] = acc1[v];
        }
        __syncthreads();

        // ---- elementwise cell update: 16 rows x 32 cols per block ----
        const float* xbase = Xpre + (size_t)t * BATCH * GDIM;
#pragma unroll
        for (int i = 0; i < 4; ++i) {
            int idx  = tid + i * 128;            // 0..511
            int ml   = idx >> 5;                 // 0..15
            int nl   = idx & 31;                 // 0..31
            int nsub = nl >> 4;
            int nl15 = nl & 15;
            int b    = tile_m * 16 + ml;
            int n    = n_grp * 32 + nl;
            const float* xp = xbase + (size_t)b * GDIM;
            float pg = sPre[0][nsub][ml * 16 + nl15] + xp[0 * HDIM + n];
            float pi = sPre[1][nsub][ml * 16 + nl15] + xp[1 * HDIM + n];
            float pf = sPre[2][nsub][ml * 16 + nl15] + xp[2 * HDIM + n];
            float po = sPre[3][nsub][ml * 16 + nl15] + xp[3 * HDIM + n];
            float g  = tanh_fast(pg);
            float ig = sig_fast(pi);
            float fg = sig_fast(pf);
            float og = sig_fast(po);
            size_t ci = (size_t)b * HDIM + n;
            float c_new = g * ig + c_st[ci] * fg;
            c_st[ci] = c_new;
            float h_new = tanh_fast(c_new) * og;
            out[((size_t)t * BATCH + b) * HDIM + n] = h_new;
            h_out[ci] = f2bf(h_new);
            if (t == T_STEPS - 1) {
                size_t tail = (size_t)T_STEPS * BATCH * HDIM;
                out[tail + (size_t)n * BATCH + b] = h_new;
                out[tail + (size_t)HDIM * BATCH + (size_t)n * BATCH + b] = c_new;
            }
        }

        // ---- publish: this block finished step t (writes + reads done) ----
        __syncthreads();
        if (tid == 0) {
            __hip_atomic_store(my_flag, (unsigned)(t + 1), __ATOMIC_RELEASE,
                               __HIP_MEMORY_SCOPE_AGENT);
        }
    }
}

// ---------------------------------------------------------------------------
// Host launcher
// ---------------------------------------------------------------------------
extern "C" void kernel_launch(void* const* d_in, const int* in_sizes, int n_in,
                              void* d_out, int out_size, void* d_ws, size_t ws_size,
                              hipStream_t stream) {
    const float* embeds = (const float*)d_in[0];
    const float* Wgx = (const float*)d_in[1];
    const float* Wgh = (const float*)d_in[2];
    const float* Wix = (const float*)d_in[3];
    const float* Wih = (const float*)d_in[4];
    const float* Wfx = (const float*)d_in[5];
    const float* Wfh = (const float*)d_in[6];
    const float* Wox = (const float*)d_in[7];
    const float* Woh = (const float*)d_in[8];
    const float* bg  = (const float*)d_in[9];
    const float* bi  = (const float*)d_in[10];
    const float* bf_ = (const float*)d_in[11];
    const float* bo  = (const float*)d_in[12];
    float* out = (float*)d_out;

    // ---- workspace layout ----
    char* ws = (char*)d_ws;
    size_t off = 0;
    unsigned short* Wx_p = (unsigned short*)(ws + off); off += (size_t)GDIM * EDIM * 2;
    unsigned short* Wh_p = (unsigned short*)(ws + off); off += (size_t)GDIM * HDIM * 2;
    float* bias_p        = (float*)(ws + off);          off += (size_t)GDIM * 4;
    off = (off + 255) & ~(size_t)255;
    unsigned short* Xbf  = (unsigned short*)(ws + off); off += (size_t)T_STEPS * BATCH * EDIM * 2;
    float* Xpre          = (float*)(ws + off);          off += (size_t)T_STEPS * BATCH * GDIM * 4;
    unsigned short* hb0  = (unsigned short*)(ws + off); off += (size_t)BATCH * HDIM * 2;
    unsigned short* hb1  = (unsigned short*)(ws + off); off += (size_t)BATCH * HDIM * 2;
    float* c_st          = (float*)(ws + off);          off += (size_t)BATCH * HDIM * 4;
    off = (off + 255) & ~(size_t)255;
    unsigned int* flags  = (unsigned int*)(ws + off);   off += RBLK * 4;
    (void)ws_size; (void)in_sizes; (void)n_in; (void)out_size;

    pack_params<<<4096, 256, 0, stream>>>(Wgx, Wgh, Wix, Wih, Wfx, Wfh, Wox, Woh,
                                          bg, bi, bf_, bo, Wx_p, Wh_p, bias_p);
    convert_embeds<<<8192, 256, 0, stream>>>(embeds, Xbf);
    init_state<<<(BATCH * HDIM + 255) / 256, 256, 0, stream>>>(hb0, c_st, flags);
    xproj_gemm<<<dim3(64, 256), 256, 0, stream>>>(Xbf, Wx_p, bias_p, Xpre);
    lstm_recurrence<<<RBLK, 128, 0, stream>>>(hb0, hb1, c_st, Wh_p, Xpre, out, flags);
}
